// DynamicConv_77695958385440
// MI455X (gfx1250) — matrix operand order repaired and checked by reading, not compiled, not run on hardware
//
#include <hip/hip_runtime.h>

// ---------------------------------------------------------------------------
// DynamicConv for MI455X (gfx1250, wave32).
//   B=32, C=256, H=W=64, K=4, RED=4, KS=3.
// Pipeline:
//   1) gap_kernel   : g[b,c] = mean_{h,w} x           (HBM read 128 MiB)
//   2) attn_kernel  : fc1 (WMMA f32) -> relu -> fc2 -> softmax -> combine
//                     w_eff[b,c,9] = sum_k attn[b,k]*conv_w[k,c,9]
//   3) conv_kernel  : depthwise 3x3 with per-(b,c) kernel via LDS tile
//                     (x likely L2-resident from pass 1; NT stores for out)
// ---------------------------------------------------------------------------

#define B_  32
#define C_  256
#define H_  64
#define W_  64
#define PLANE (H_ * W_)        // 4096
#define NP  (B_ * C_)          // 8192 planes
#define HID 64                 // C/RED

typedef __attribute__((ext_vector_type(2)))  float    v2f;
typedef __attribute__((ext_vector_type(8)))  float    v8f;
typedef __attribute__((ext_vector_type(16))) _Float16 v16h;

// ---------------- 1) global average pool: one block per (b,c) plane --------
__global__ __launch_bounds__(256) void gap_kernel(const float* __restrict__ x,
                                                  float* __restrict__ g) {
  __shared__ float sdata[256];
  const int plane = blockIdx.x;                       // b*C + c
  const float4* p = (const float4*)(x + (size_t)plane * PLANE);
  const int t = threadIdx.x;
  float s = 0.f;
#pragma unroll
  for (int i = 0; i < 4; ++i) {                       // 1024 float4 / 256 thr
    float4 v = p[t + i * 256];
    s += (v.x + v.y) + (v.z + v.w);
  }
  sdata[t] = s;
  __syncthreads();
  for (int off = 128; off > 0; off >>= 1) {
    if (t < off) sdata[t] += sdata[t + off];
    __syncthreads();
  }
  if (t == 0) g[plane] = sdata[0] * (1.0f / (float)PLANE);
}

// ---------------- 2) attention branch + per-(b,c) weight combine -----------
// One block, 256 threads = 8 waves. Each wave owns one 16x16 tile of
// h = relu(g @ fc1_w^T + b1): M=32 rows (batch) x N=64 cols -> 2x4 tiles.
__global__ __launch_bounds__(256) void attn_kernel(
    const float* __restrict__ g,       // [32,256]
    const float* __restrict__ conv_w,  // [4,256,1,3,3]
    const float* __restrict__ fc1_w,   // [64,256]
    const float* __restrict__ fc1_b,   // [64]
    const float* __restrict__ fc2_w,   // [4,64]
    const float* __restrict__ fc2_b,   // [4]
    float* __restrict__ w_eff) {       // [32*256, 9]
  __shared__ float h_lds[B_ * HID];    // 32x64
  __shared__ float attn_lds[B_ * 4];

  const int tid  = threadIdx.x;
  const int lane = tid & 31;
  const int wv   = tid >> 5;           // 0..7
  const int mt   = wv >> 2;            // M tile: 0..1
  const int nt   = wv & 3;             // N tile: 0..3
  const int hf   = lane >> 4;          // lane half (0/1)
  const int lm   = lane & 15;
  const int arow = mt * 16 + lm;       // batch row this lane feeds to A
  const int bcol = nt * 16 + lm;       // output col this lane feeds to B

  v8f c8 = {};

#if __has_builtin(__builtin_amdgcn_wmma_f32_16x16x4_f32)
  // Native fp32 WMMA: A 16x4, B 4x16, f32 accum. ISA layout:
  //   A VGPR0: K=0 (lanes 0-15) / K=2 (lanes 16-31); VGPR1: K=1 / K=3.
  //   B mirrors with N striped across lanes.
  for (int k0 = 0; k0 < C_; k0 += 4) {
    v2f a, b;
    const int ka = k0 + 2 * hf;
    a.x = g[arow * C_ + ka];
    a.y = g[arow * C_ + ka + 1];
    b.x = fc1_w[bcol * C_ + ka];       // B[k][n] = fc1_w[n][k]
    b.y = fc1_w[bcol * C_ + ka + 1];
    c8 = __builtin_amdgcn_wmma_f32_16x16x4_f32(
        false, a, false, b, (short)0, c8, false, false);
  }
#else
  // Fallback: codegen-confirmed f16 WMMA (16x16x32), f32 accumulate.
  for (int k0 = 0; k0 < C_; k0 += 32) {
    v16h a, b;
#pragma unroll
    for (int e = 0; e < 16; ++e) {
      const int r = e >> 1, p = e & 1;
      const int ka = (r < 4) ? (2 * r + p + 8 * hf)
                             : (16 + 2 * (r - 4) + p + 8 * hf);
      a[e] = (_Float16)g[arow * C_ + k0 + ka];
      const int kb = (e < 8) ? (e + 8 * hf) : (16 + (e - 8) + 8 * hf);
      b[e] = (_Float16)fc1_w[bcol * C_ + k0 + kb];
    }
    c8 = __builtin_amdgcn_wmma_f32_16x16x32_f16(
        false, a, false, b, (short)0, c8, false, false);
  }
#endif

  // D layout: VGPR r -> M = r (lanes 0-15) / r+8 (lanes 16-31), N = lane%16.
  const float bias = fc1_b[bcol];
#pragma unroll
  for (int r = 0; r < 8; ++r) {
    const int row = mt * 16 + hf * 8 + r;
    const float v = c8[r] + bias;
    h_lds[row * HID + bcol] = v > 0.f ? v : 0.f;
  }
  __syncthreads();

  // fc2 + softmax: one lane per batch row.
  if (tid < B_) {
    float acc[4];
#pragma unroll
    for (int k = 0; k < 4; ++k) acc[k] = fc2_b[k];
    for (int j = 0; j < HID; ++j) {
      const float hv = h_lds[tid * HID + j];
#pragma unroll
      for (int k = 0; k < 4; ++k) acc[k] += hv * fc2_w[k * HID + j];
    }
    float m = acc[0];
#pragma unroll
    for (int k = 1; k < 4; ++k) m = acc[k] > m ? acc[k] : m;
    float e[4], s = 0.f;
#pragma unroll
    for (int k = 0; k < 4; ++k) { e[k] = __expf(acc[k] - m); s += e[k]; }
    const float inv = 1.0f / s;
#pragma unroll
    for (int k = 0; k < 4; ++k) attn_lds[tid * 4 + k] = e[k] * inv;
  }
  __syncthreads();

  // Combine the K=4 depthwise kernels into one per-(b,c) 3x3 kernel.
  for (int idx = tid; idx < NP; idx += 256) {
    const int b = idx >> 8;
    const int c = idx & 255;
    const float a0 = attn_lds[b * 4 + 0];
    const float a1 = attn_lds[b * 4 + 1];
    const float a2 = attn_lds[b * 4 + 2];
    const float a3 = attn_lds[b * 4 + 3];
#pragma unroll
    for (int t9 = 0; t9 < 9; ++t9) {
      const float w = a0 * conv_w[(0 * C_ + c) * 9 + t9]
                    + a1 * conv_w[(1 * C_ + c) * 9 + t9]
                    + a2 * conv_w[(2 * C_ + c) * 9 + t9]
                    + a3 * conv_w[(3 * C_ + c) * 9 + t9];
      w_eff[(size_t)idx * 9 + t9] = w;
    }
  }
}

// ---------------- 3) depthwise 3x3 with per-plane kernel -------------------
#define TPITCH 68                      // 66 cols padded to 68 (bank spread)
__global__ __launch_bounds__(256) void conv_kernel(
    const float* __restrict__ x,
    const float* __restrict__ w_eff,
    float* __restrict__ out) {
  __shared__ float tile[66 * TPITCH]; // ~17.9 KB
  const int plane = blockIdx.x;        // b*C + c
  const float* p = x + (size_t)plane * PLANE;
  const int tid = threadIdx.x;

  // Stage 66x66 halo tile (zero padding at borders) into LDS.
  for (int idx = tid; idx < 66 * 66; idx += 256) {
    const int r = idx / 66;
    const int c = idx - r * 66;
    const int gr = r - 1, gc = c - 1;
    float v = 0.f;
    if ((unsigned)gr < 64u && (unsigned)gc < 64u) v = p[gr * W_ + gc];
    tile[r * TPITCH + c] = v;
  }

  const float* wp = w_eff + (size_t)plane * 9;
  const float w0 = wp[0], w1 = wp[1], w2 = wp[2];
  const float w3 = wp[3], w4 = wp[4], w5 = wp[5];
  const float w6 = wp[6], w7 = wp[7], w8 = wp[8];
  __syncthreads();

  float* po = out + (size_t)plane * PLANE;
#pragma unroll
  for (int i = 0; i < 16; ++i) {
    const int oi = i * 256 + tid;
    const int r = oi >> 6;
    const int c = oi & 63;
    const float* t0 = &tile[r * TPITCH + c];
    const float v = w0 * t0[0]            + w1 * t0[1]            + w2 * t0[2]
                  + w3 * t0[TPITCH]       + w4 * t0[TPITCH + 1]   + w5 * t0[TPITCH + 2]
                  + w6 * t0[2 * TPITCH]   + w7 * t0[2 * TPITCH + 1] + w8 * t0[2 * TPITCH + 2];
    __builtin_nontemporal_store(v, &po[oi]);   // out never re-read: keep L2 for x
  }
}

// ---------------------------------------------------------------------------
extern "C" void kernel_launch(void* const* d_in, const int* in_sizes, int n_in,
                              void* d_out, int out_size, void* d_ws, size_t ws_size,
                              hipStream_t stream) {
  const float* x      = (const float*)d_in[0];
  const float* conv_w = (const float*)d_in[1];
  const float* fc1_w  = (const float*)d_in[2];
  const float* fc1_b  = (const float*)d_in[3];
  const float* fc2_w  = (const float*)d_in[4];
  const float* fc2_b  = (const float*)d_in[5];
  float* out = (float*)d_out;

  float* g     = (float*)d_ws;                         // 8192 floats (32 KB)
  float* w_eff = (float*)((char*)d_ws + 32 * 1024);    // 73728 floats (288 KB)

  gap_kernel<<<NP, 256, 0, stream>>>(x, g);
  attn_kernel<<<1, 256, 0, stream>>>(g, conv_w, fc1_w, fc1_b, fc2_w, fc2_b, w_eff);
  conv_kernel<<<NP, 256, 0, stream>>>(x, w_eff, out);
}
